// Encoder_GNN_v_weighted_46815143526427
// MI455X (gfx1250) — matrix operand order repaired and checked by reading, not compiled, not run on hardware
//
#include <hip/hip_runtime.h>
#include <hip/hip_bf16.h>

// ---------------------------------------------------------------------------
// GraphSAGE (2x SAGEConv mean + linear out) for MI455X / gfx1250.
// GEMMs run on v_wmma_f32_16x16x32_bf16 (f32 accumulate); aggregation is an
// edge-parallel float4 gather + global_atomic_add_f32 scatter (L2-resident).
// ---------------------------------------------------------------------------

typedef __attribute__((ext_vector_type(16))) __bf16 v16bf;
typedef __attribute__((ext_vector_type(8)))  __bf16 v8bf;
typedef __attribute__((ext_vector_type(8)))  float  v8f;

typedef unsigned short bfu;   // raw bf16 storage

__device__ __forceinline__ bfu f2bf(float f) {
  unsigned int u = __float_as_uint(f);
  unsigned int r = 0x7FFFu + ((u >> 16) & 1u);   // round to nearest even
  return (bfu)((u + r) >> 16);
}

// ---------------------------------------------------------------------------
__global__ void zero_f32_k(float* __restrict__ p, long long n) {
  long long i = (long long)blockIdx.x * blockDim.x + threadIdx.x;
  long long stride = (long long)gridDim.x * blockDim.x;
  for (; i < n; i += stride) p[i] = 0.0f;
}

__global__ void degree_k(const int* __restrict__ dst, float* __restrict__ cnt, int E) {
  int e = blockIdx.x * blockDim.x + threadIdx.x;
  if (e < E) atomicAdd(&cnt[dst[e]], 1.0f);
}

// one thread per (edge, 4-float group): gather x[src] row, atomic-add into agg[dst]
__global__ void scatter_k(const float* __restrict__ X, const int* __restrict__ src,
                          const int* __restrict__ dst, float* __restrict__ agg,
                          long long total, int F, int lgFv) {
  long long tid = (long long)blockIdx.x * blockDim.x + threadIdx.x;
  if (tid >= total) return;
  int Fv = F >> 2;
  int e = (int)(tid >> lgFv);
  int g = (int)(tid & (Fv - 1));
  int s = src[e], d = dst[e];
  const float4 v = *(const float4*)(X + (size_t)s * F + 4 * g);
  float* o = agg + (size_t)d * F + 4 * g;
  atomicAdd(o + 0, v.x); atomicAdd(o + 1, v.y);
  atomicAdd(o + 2, v.z); atomicAdd(o + 3, v.w);
}

// A[p*ldA + colOff + f] = bf16( agg[p*F + f] / max(cnt[p], 1) )
__global__ void finalize_mean_k(const float* __restrict__ agg, const float* __restrict__ cnt,
                                bfu* __restrict__ A, int ldA, int colOff, int F,
                                long long total) {
  long long tid = (long long)blockIdx.x * blockDim.x + threadIdx.x;
  if (tid >= total) return;
  long long p = tid / F;
  int f = (int)(tid - p * F);
  float c = fmaxf(cnt[p], 1.0f);
  A[p * ldA + colOff + f] = f2bf(agg[tid] / c);
}

// A[p*ldA + colOff + f] = bf16( p < Nreal ? src[p*F + f] : 0 )
__global__ void cast_cat_k(const float* __restrict__ srcm, bfu* __restrict__ A,
                           int ldA, int colOff, int F, long long Nreal, long long total) {
  long long tid = (long long)blockIdx.x * blockDim.x + threadIdx.x;
  if (tid >= total) return;
  long long p = tid / F;
  int f = (int)(tid - p * F);
  float v = (p < Nreal) ? srcm[p * F + f] : 0.0f;
  A[p * ldA + colOff + f] = f2bf(v);
}

// Wt[n*K + k] = bf16( k < K1 ? Wa[k*Nd + n] : Wb[(k-K1)*Nd + n] )   (K = K1+K2)
__global__ void prep_w_k(const float* __restrict__ Wa, const float* __restrict__ Wb,
                         bfu* __restrict__ Wt, int K1, int K2, int Nd) {
  int K = K1 + K2;
  long long total = (long long)Nd * K;
  long long tid = (long long)blockIdx.x * blockDim.x + threadIdx.x;
  if (tid >= total) return;
  int n = (int)(tid / K);
  int k = (int)(tid - (long long)n * K);
  float v = (k < K1) ? Wa[(size_t)k * Nd + n] : Wb[(size_t)(k - K1) * Nd + n];
  Wt[(size_t)n * K + k] = f2bf(v);
}

// ---------------------------------------------------------------------------
// WMMA GEMM:  out[M x Nd] = act( A[M x K](bf16) * Wt[Nd x K]^T(bf16) + bias )
// A row-major [M][K]; Wt holds B pre-transposed, row-major [Nd][K].
// Block: 256 thr = 8 waves; wave w -> rows m0 = bx*128 + w*16; by -> 16 cols.
// Layouts as used here (A and B share a contiguous k order, not the ISA table order):
//   A 16x32 frag : lane L: row = L&15;  two 16B chunks at k + (L>=16?16:0) and +8.
//   B 32x16 frag : lane L: col = L&15;  one 32B chunk  at k + (L>=16?16:0).
//   C/D 16x16    : lane L: col = L&15;  elem j -> row j + (L>=16?8:0).
// ---------------------------------------------------------------------------
__global__ void __launch_bounds__(256)
gemm_wmma_k(const bfu* __restrict__ A, const bfu* __restrict__ Wt,
            const float* __restrict__ bias,
            float* __restrict__ outF, bfu* __restrict__ outB,
            int K, int Nd, int Mreal, int ldF, int ldB, int colOffB, int relu) {
  const int lane = threadIdx.x & 31;
  const int wave = threadIdx.x >> 5;
  const int m0 = blockIdx.x * 128 + wave * 16;
  const int n0 = blockIdx.y * 16;
  const bool hi = lane >= 16;
  const int l15 = lane & 15;

  const bfu* Arow = A  + (size_t)(m0 + l15) * K + (hi ? 16 : 0);
  const bfu* Brow = Wt + (size_t)(n0 + l15) * K + (hi ? 16 : 0);

  v8f acc = {};
  for (int k = 0; k < K; k += 32) {
    v8bf alo = *(const v8bf*)(Arow + k);
    v8bf ahi = *(const v8bf*)(Arow + k + 8);
    v16bf a = __builtin_shufflevector(alo, ahi, 0, 1, 2, 3, 4, 5, 6, 7,
                                                8, 9, 10, 11, 12, 13, 14, 15);
    v16bf b = *(const v16bf*)(Brow + k);
    acc = __builtin_amdgcn_wmma_f32_16x16x32_bf16(
        /*neg_a=*/false, a, /*neg_b=*/false, b,
        /*c_mod=*/(short)0, acc, /*reuse_a=*/false, /*reuse_b=*/false);
  }

  const int n = n0 + l15;
  const float bv = bias[n];
#pragma unroll
  for (int j = 0; j < 8; ++j) {
    int m = m0 + j + (hi ? 8 : 0);
    float v = acc[j] + bv;
    if (relu) v = fmaxf(v, 0.0f);
    if (outF != nullptr && m < Mreal) outF[(size_t)m * ldF + n] = v;
    if (outB != nullptr)              outB[(size_t)m * ldB + colOffB + n] = f2bf(v);
  }
}

// ---------------------------------------------------------------------------
extern "C" void kernel_launch(void* const* d_in, const int* in_sizes, int n_in,
                              void* d_out, int out_size, void* d_ws, size_t ws_size,
                              hipStream_t stream) {
  (void)n_in; (void)out_size; (void)ws_size;

  const float* x   = (const float*)d_in[0];
  const int*   ei  = (const int*)d_in[1];
  const float* W1l = (const float*)d_in[2];
  const float* b1  = (const float*)d_in[3];
  const float* W1r = (const float*)d_in[4];
  const float* W2l = (const float*)d_in[5];
  const float* b2  = (const float*)d_in[6];
  const float* W2r = (const float*)d_in[7];
  const float* Wo  = (const float*)d_in[8];
  const float* bo  = (const float*)d_in[9];
  float* out = (float*)d_out;

  const int HID = in_sizes[3];                  // 256
  const int FIN = in_sizes[2] / HID;            // 128
  const int OUT = in_sizes[9];                  // 128
  const long long N  = (long long)in_sizes[0] / FIN;   // 50000
  const long long E  = (long long)in_sizes[1] / 2;     // 1.6M
  const long long Mp = ((N + 127) / 128) * 128;        // 50048 (padded rows)

  const int* src = ei;
  const int* dst = ei + E;

  // ---- workspace carve-out (256B aligned) ----
  char* ws = (char*)d_ws;
  size_t off = 0;
  auto carve = [&](size_t bytes) -> void* {
    void* p = ws + off;
    off = (off + bytes + 255) & ~(size_t)255;
    return p;
  };
  float* cnt = (float*)carve((size_t)Mp * 4);
  float* agg = (float*)carve((size_t)Mp * HID * 4);          // reused both layers
  bfu*   A1  = (bfu*)  carve((size_t)Mp * (2 * FIN) * 2);    // [mean(x)|x]      bf16
  float* H1  = (float*)carve((size_t)Mp * HID * 4);          // layer-1 out      f32
  bfu*   A2  = (bfu*)  carve((size_t)Mp * (2 * HID) * 2);    // [mean(h1)|h1]    bf16
  bfu*   A3  = (bfu*)  carve((size_t)Mp * HID * 2);          // layer-2 out      bf16
  bfu*   Wt1 = (bfu*)  carve((size_t)HID * (2 * FIN) * 2);   // [W1l;W1r]^T      bf16
  bfu*   Wt2 = (bfu*)  carve((size_t)HID * (2 * HID) * 2);   // [W2l;W2r]^T      bf16
  bfu*   Wto = (bfu*)  carve((size_t)OUT * HID * 2);         // Wo^T             bf16

  auto nb = [](long long n) { return (unsigned)((n + 255) / 256); };
  const dim3 B256(256);

  // ---- degree counts (shared by both layers) ----
  zero_f32_k<<<dim3(nb(Mp) < 4096u ? nb(Mp) : 4096u), B256, 0, stream>>>(cnt, Mp);
  degree_k<<<dim3(nb(E)), B256, 0, stream>>>(dst, cnt, (int)E);

  // ---- pre-transpose weights to bf16 [Nd][K] ----
  prep_w_k<<<dim3(nb((long long)HID * 2 * FIN)), B256, 0, stream>>>(W1l, W1r, Wt1, FIN, FIN, HID);
  prep_w_k<<<dim3(nb((long long)HID * 2 * HID)), B256, 0, stream>>>(W2l, W2r, Wt2, HID, HID, HID);
  prep_w_k<<<dim3(nb((long long)OUT * HID)),     B256, 0, stream>>>(Wo,  Wo,  Wto, HID, 0,   OUT);

  // ================= Layer 1 =================
  zero_f32_k<<<dim3(16384), B256, 0, stream>>>(agg, Mp * (long long)FIN);
  {
    int lg = 0; while ((1 << lg) < (FIN >> 2)) ++lg;          // log2(F/4)
    long long total = E * (FIN >> 2);
    scatter_k<<<dim3(nb(total)), B256, 0, stream>>>(x, src, dst, agg, total, FIN, lg);
  }
  finalize_mean_k<<<dim3(nb(Mp * FIN)), B256, 0, stream>>>(agg, cnt, A1, 2 * FIN, 0, FIN, Mp * FIN);
  cast_cat_k<<<dim3(nb(Mp * FIN)), B256, 0, stream>>>(x, A1, 2 * FIN, FIN, FIN, N, Mp * FIN);

  // h1 = relu(A1 @ Wt1^T + b1) -> H1 (f32) and A2[:, HID:2*HID] (bf16)
  gemm_wmma_k<<<dim3((unsigned)(Mp / 128), HID / 16), B256, 0, stream>>>(
      A1, Wt1, b1, H1, A2, 2 * FIN, HID, (int)Mp, HID, 2 * HID, HID, 1);

  // ================= Layer 2 =================
  zero_f32_k<<<dim3(16384), B256, 0, stream>>>(agg, Mp * (long long)HID);
  {
    int lg = 0; while ((1 << lg) < (HID >> 2)) ++lg;
    long long total = E * (HID >> 2);
    scatter_k<<<dim3(nb(total)), B256, 0, stream>>>(H1, src, dst, agg, total, HID, lg);
  }
  finalize_mean_k<<<dim3(nb(Mp * HID)), B256, 0, stream>>>(agg, cnt, A2, 2 * HID, 0, HID, Mp * HID);

  // h2 = relu(A2 @ Wt2^T + b2) -> A3 (bf16)
  gemm_wmma_k<<<dim3((unsigned)(Mp / 128), HID / 16), B256, 0, stream>>>(
      A2, Wt2, b2, nullptr, A3, 2 * HID, HID, (int)Mp, 0, HID, 0, 1);

  // ================= Output =================
  // out = A3 @ Wto^T + bo   (f32, row-guarded to N)
  gemm_wmma_k<<<dim3((unsigned)(Mp / 128), OUT / 16), B256, 0, stream>>>(
      A3, Wto, bo, out, nullptr, HID, OUT, (int)N, OUT, 0, 0, 0);
}
